// MaskRCNNFPNFeatureExtractor_36996848287889
// MI455X (gfx1250) — matrix operand order repaired and checked by reading, not compiled, not run on hardware
//
#include <hip/hip_runtime.h>
#include <hip/hip_bf16.h>

// ---------------------------------------------------------------------------
// MaskRCNN FPN feature extractor, fused for MI455X (gfx1250, wave32, WMMA).
//  - 1 workgroup (256 thr = 8 waves) per ROI
//  - ROIAlign -> LDS (bf16, channel-last, padded stride 264 for bank-free WMMA
//    fragment loads)
//  - 4x conv3x3+ReLU as implicit GEMM on v_wmma_f32_16x16x32_bf16, ping-pong
//    between two 132KB LDS activation buffers (all intermediates on-chip)
//  - weights repacked once per launch into the exact per-lane WMMA B layout
//  - A-fragment LDS addresses advanced by constant byte deltas per tap so the
//    k-loop has no multiplies (fewer VALU ops / WMMA-hazard v_nops)
// ---------------------------------------------------------------------------

typedef __attribute__((ext_vector_type(16))) __bf16 v16bf;
typedef __attribute__((ext_vector_type(8)))  float  v8f;

union ABR { uint4 u[2]; v16bf v; };

#define RES        14
#define CH         256
#define PIXSTRIDE  264              // 256 ch + 8 pad halfwords -> bank-conflict-free
#define PIXB       (PIXSTRIDE * 2)            // bytes per pixel row (528)
#define BUF_HW     (256 * PIXSTRIDE)          // halfwords per activation buffer
#define BUF_BYTES  (BUF_HW * 2)               // 135,168 B
#define TAB_OFF    (2 * BUF_BYTES)            // sample table after two buffers
#define NSAMP      (28 * 28)                  // 784 bilinear samples per ROI
#define SMEM_BYTES (TAB_OFF + NSAMP * 32)     // 295,424 B  (< 320 KB / WGP)
#define WL         (16 * 72 * 32 * 16)        // packed halfwords per conv layer
#define NPACK      (4 * WL)                   // 2,359,296 halfwords = 4.72 MB

__device__ __forceinline__ unsigned short f2bf(float f) {
    unsigned int u = __float_as_uint(f);
    u += 0x7FFFu + ((u >> 16) & 1u);          // round-to-nearest-even
    return (unsigned short)(u >> 16);
}

// ---------------------------------------------------------------------------
// Weight repack:  bp[layer][n_tile(16)][k_step(72)][lane(32)][j(16)]  (bf16)
// K is reordered as K = tap*256 + ci (tap = kh*3+kw) so that each 32-wide
// K-step is a contiguous channel run.  Per-lane j run matches the ISA B
// layout: lanes 0-15 hold K 0..7,16..23 of the step, lanes 16-31 hold K 8..15,24..31.
// ---------------------------------------------------------------------------
__global__ void pack_weights_kernel(const float* __restrict__ w1,
                                    const float* __restrict__ w2,
                                    const float* __restrict__ w3,
                                    const float* __restrict__ w4,
                                    unsigned short* __restrict__ bp) {
    unsigned int idx = blockIdx.x * 256u + threadIdx.x;
    if (idx >= (unsigned)NPACK) return;
    int j    = idx & 15;
    int lane = (idx >> 4) & 31;
    int ks   = (int)((idx >> 9) % 72u);
    int lnt  = (int)((idx >> 9) / 72u);
    int nt   = lnt & 15;
    int layer = lnt >> 4;
    int tap = ks >> 3;                         // kh*3+kw
    int ci  = (ks & 7) * 32 + ((lane >> 4) << 3) + (j & 7) + ((j >> 3) << 4);
    int n   = nt * 16 + (lane & 15);
    long widx = ((long)n * 256 + ci) * 9 + tap;   // OIHW flat
    const float* w = (layer == 0) ? w1 : (layer == 1) ? w2 : (layer == 2) ? w3 : w4;
    bp[idx] = f2bf(w[widx]);
}

// ---------------------------------------------------------------------------
// One M-chunk of the implicit-GEMM conv for one wave.
//   A: activations from LDS (M = spatial pixel, K = (tap,ci))
//   B: packed weights from global (N = out channel)
//   acc: MCH m-tiles x 2 n-tiles of 16x16 f32
// ---------------------------------------------------------------------------
template <int MCH, bool FINAL>
__device__ __forceinline__ void conv_chunk(const unsigned short* __restrict__ actIn,
                                           unsigned short* __restrict__ actOut,
                                           float* __restrict__ out,
                                           const float* __restrict__ bias,
                                           const unsigned short* __restrict__ wl,
                                           int lane, int nTile0, int mTile0, int roi) {
    v8f acc[2][MCH] = {};

    const int mloc = lane & 15;
    // byte offset of each m-row's tap-(0,0) input pixel inside the LDS buffer
    int pbB[MCH];
#pragma unroll
    for (int i = 0; i < MCH; ++i) {
        int m = (mTile0 + i) * 16 + mloc;
        m = m < 195 ? m : 195;                 // clamp pad rows (outputs masked later)
        pbB[i] = ((m / RES) * 16 + (m % RES)) * PIXB;
    }
    const char* actB = (const char*)actIn;
    const int ciHalfB = (lane >> 4) << 4;      // A-fragment lane-half K offset, bytes
    const unsigned short* bPtr = wl + ((size_t)nTile0 * 72 * 32 + lane) * 16;

    for (int tap = 0; tap < 9; ++tap) {
        for (int cs = 0; cs < 8; ++cs) {
            const int ks = tap * 8 + cs;
            // two B fragments: 32 B contiguous per lane, coalesced per wave
            ABR b0, b1;
            const uint4* bp0 = (const uint4*)(bPtr + ks * 512);
            const uint4* bp1 = (const uint4*)(bPtr + 72 * 512 + ks * 512);
            b0.u[0] = bp0[0]; b0.u[1] = bp0[1];
            b1.u[0] = bp1[0]; b1.u[1] = bp1[1];

            const int ciAB = cs * 64 + ciHalfB;   // byte offset of K-run within pixel
#pragma unroll
            for (int i = 0; i < MCH; ++i) {
                ABR a;   // ISA 16-bit A layout: runs K{0..7} and K{16..23} per half
                a.u[0] = *(const uint4*)(actB + pbB[i] + ciAB);
                a.u[1] = *(const uint4*)(actB + pbB[i] + ciAB + 32);
                acc[0][i] = __builtin_amdgcn_wmma_f32_16x16x32_bf16(
                    false, a.v, false, b0.v, (short)0, acc[0][i], false, false);
                acc[1][i] = __builtin_amdgcn_wmma_f32_16x16x32_bf16(
                    false, a.v, false, b1.v, (short)0, acc[1][i], false, false);
            }
        }
        // advance to next tap: +1 col, or +1 row / -2 cols on row wrap
        const int d = (tap % 3 == 2) ? (14 * PIXB) : PIXB;
#pragma unroll
        for (int i = 0; i < MCH; ++i) pbB[i] += d;
    }

    // ---- epilogue: bias + ReLU, scatter per C/D layout (M = r + 8*(lane>>4))
    const int mHalf = (lane >> 4) << 3;
#pragma unroll
    for (int ntl = 0; ntl < 2; ++ntl) {
        const int n = (nTile0 + ntl) * 16 + (lane & 15);
        const float bv = bias[n];
#pragma unroll
        for (int i = 0; i < MCH; ++i) {
#pragma unroll
            for (int r = 0; r < 8; ++r) {
                const int m = (mTile0 + i) * 16 + mHalf + r;
                if (m < RES * RES) {
                    float v = fmaxf(acc[ntl][i][r] + bv, 0.f);
                    if (FINAL) {
                        out[((size_t)roi * CH + n) * (RES * RES) + m] = v;
                    } else {
                        actOut[((m / RES + 1) * 16 + (m % RES) + 1) * PIXSTRIDE + n] =
                            f2bf(v);
                    }
                }
            }
        }
    }
}

template <bool FINAL>
__device__ __forceinline__ void conv_layer(const unsigned short* actIn,
                                           unsigned short* actOut, float* out,
                                           const float* bias,
                                           const unsigned short* wl,
                                           int tid, int roi) {
    const int lane = tid & 31;
    const int nTile0 = (tid >> 5) * 2;         // wave -> 2 of 16 n-tiles
    conv_chunk<5, FINAL>(actIn, actOut, out, bias, wl, lane, nTile0, 0, roi);
    conv_chunk<4, FINAL>(actIn, actOut, out, bias, wl, lane, nTile0, 5, roi);
    conv_chunk<4, FINAL>(actIn, actOut, out, bias, wl, lane, nTile0, 9, roi);
}

// ---------------------------------------------------------------------------
// Fused per-ROI kernel: ROIAlign -> 4x (conv3x3 + ReLU), all in LDS.
// ---------------------------------------------------------------------------
__global__ void __launch_bounds__(256, 1)
MaskRCNN_FPN_fused_kernel(const float* __restrict__ f0, const float* __restrict__ f1,
                          const float* __restrict__ f2, const float* __restrict__ f3,
                          const float* __restrict__ rois,
                          const float* __restrict__ b1, const float* __restrict__ b2,
                          const float* __restrict__ b3, const float* __restrict__ b4,
                          const unsigned short* __restrict__ wpack,
                          float* __restrict__ out) {
    extern __shared__ unsigned char smem[];
    unsigned short* bufA = (unsigned short*)smem;
    unsigned short* bufB = bufA + BUF_HW;
    int4*   tOff = (int4*)(smem + TAB_OFF);
    float4* tW   = (float4*)(smem + TAB_OFF + NSAMP * 16);

    const int tid = threadIdx.x;
    const int roi = blockIdx.x;

    // ---- per-ROI scalars (computed redundantly by every thread) ----
    const float x1 = rois[roi * 5 + 1], y1v = rois[roi * 5 + 2];
    const float x2 = rois[roi * 5 + 3], y2v = rois[roi * 5 + 4];
    const int bi = (int)rois[roi * 5 + 0];
    const float area = fmaxf((x2 - x1) * (y2v - y1v), 1e-6f);
    float lf = floorf(4.f + log2f(sqrtf(area) / 224.f + 1e-6f));
    lf = fminf(fmaxf(lf, 2.f), 5.f);
    const int lvl = (int)lf - 2;
    const float SCv[4] = {0.25f, 0.125f, 0.0625f, 0.03125f};
    const int Hs[4] = {200, 100, 50, 25};
    const int Ws[4] = {304, 152, 76, 38};
    const float sc = SCv[lvl];
    const int H = Hs[lvl], W = Ws[lvl];
    const float Hf = (float)H, Wf = (float)W;
    const float* fptr = (lvl == 0) ? f0 : (lvl == 1) ? f1 : (lvl == 2) ? f2 : f3;
    const float x1s = x1 * sc, y1s = y1v * sc;
    const float rw = fmaxf(x2 * sc - x1s, 1.f);
    const float rh = fmaxf(y2v * sc - y1s, 1.f);
    const float sx_step = rw * (1.f / RES), sy_step = rh * (1.f / RES);

    // ---- zero both activation buffers (halos must be 0 for padded conv) ----
    {
        uint4 z = {0u, 0u, 0u, 0u};
        uint4* p = (uint4*)smem;
        for (int i = tid; i < (2 * BUF_BYTES) / 16; i += 256) p[i] = z;
    }

    // ---- precompute 28x28 bilinear sample table (shared by all channels) ----
    for (int s = tid; s < NSAMP; s += 256) {
        const int sy = s / 28, sx = s % 28;
        const float gy = (float)(sy >> 1) + 0.25f + 0.5f * (float)(sy & 1);
        const float gx = (float)(sx >> 1) + 0.25f + 0.5f * (float)(sx & 1);
        float y = y1s + gy * sy_step;
        float x = x1s + gx * sx_step;
        const bool valid = (y > -1.f) && (y < Hf) && (x > -1.f) && (x < Wf);
        y = fminf(fmaxf(y, 0.f), Hf - 1.f);
        x = fminf(fmaxf(x, 0.f), Wf - 1.f);
        const float y0f = floorf(y), x0f = floorf(x);
        const float ly = y - y0f, lx = x - x0f;
        const int y0 = (int)y0f, x0 = (int)x0f;
        const int y1i = min(y0 + 1, H - 1), x1i = min(x0 + 1, W - 1);
        const float wv = valid ? 0.25f : 0.f;   // fold 2x2 subsample mean
        const float hy = 1.f - ly, hx = 1.f - lx;
        tOff[s] = make_int4(y0 * W + x0, y0 * W + x1i, y1i * W + x0, y1i * W + x1i);
        tW[s]   = make_float4(hy * hx * wv, hy * lx * wv, ly * hx * wv, ly * lx * wv);
    }
    __syncthreads();

    // ---- ROIAlign: thread = channel, gather 196 pixels into bufA interior ----
    {
        const float* fb = fptr + (size_t)(bi * CH + tid) * (size_t)(H * W);
        for (int p = 0; p < RES * RES; ++p) {
            const int py = p / RES, px = p % RES;
            float acc = 0.f;
#pragma unroll
            for (int sub = 0; sub < 4; ++sub) {
                const int s = (py * 2 + (sub >> 1)) * 28 + px * 2 + (sub & 1);
                const int4 o = tOff[s];
                const float4 w = tW[s];
                acc += w.x * fb[o.x] + w.y * fb[o.y] + w.z * fb[o.z] + w.w * fb[o.w];
            }
            bufA[((py + 1) * 16 + px + 1) * PIXSTRIDE + tid] = f2bf(acc);
        }
    }
    __syncthreads();

    // ---- 4 fused conv3x3+ReLU layers, ping-pong in LDS ----
    conv_layer<false>(bufA, bufB, nullptr, b1, wpack + 0 * WL, tid, roi);
    __syncthreads();
    conv_layer<false>(bufB, bufA, nullptr, b2, wpack + 1 * WL, tid, roi);
    __syncthreads();
    conv_layer<false>(bufA, bufB, nullptr, b3, wpack + 2 * WL, tid, roi);
    __syncthreads();
    conv_layer<true>(bufB, nullptr, out, b4, wpack + 3 * WL, tid, roi);
}

// ---------------------------------------------------------------------------
extern "C" void kernel_launch(void* const* d_in, const int* in_sizes, int n_in,
                              void* d_out, int out_size, void* d_ws, size_t ws_size,
                              hipStream_t stream) {
    const float* f0   = (const float*)d_in[0];
    const float* f1   = (const float*)d_in[1];
    const float* f2   = (const float*)d_in[2];
    const float* f3   = (const float*)d_in[3];
    const float* rois = (const float*)d_in[4];
    const float* w1 = (const float*)d_in[5];  const float* b1 = (const float*)d_in[6];
    const float* w2 = (const float*)d_in[7];  const float* b2 = (const float*)d_in[8];
    const float* w3 = (const float*)d_in[9];  const float* b3 = (const float*)d_in[10];
    const float* w4 = (const float*)d_in[11]; const float* b4 = (const float*)d_in[12];
    unsigned short* wpack = (unsigned short*)d_ws;   // 4.72 MB of ws
    float* out = (float*)d_out;

    // repack conv weights into the WMMA B-fragment layout (every call; cheap)
    pack_weights_kernel<<<NPACK / 256, 256, 0, stream>>>(w1, w2, w3, w4, wpack);

    // allow >64KB dynamic LDS (320KB/WGP on gfx1250)
    (void)hipFuncSetAttribute((const void*)MaskRCNN_FPN_fused_kernel,
                              hipFuncAttributeMaxDynamicSharedMemorySize, SMEM_BYTES);

    MaskRCNN_FPN_fused_kernel<<<512, 256, SMEM_BYTES, stream>>>(
        f0, f1, f2, f3, rois, b1, b2, b3, b4, wpack, out);
}
